// MultiHeadAttention_11132555231310
// MI455X (gfx1250) — matrix operand order repaired and checked by reading, not compiled, not run on hardware
//
#include <hip/hip_runtime.h>

#define DEV __device__ __forceinline__

typedef __attribute__((ext_vector_type(16))) __bf16 v16bf;
typedef __attribute__((ext_vector_type(8)))  __bf16 v8bf;
typedef __attribute__((ext_vector_type(8)))  float  v8f;

constexpr int BATCH = 4;
constexpr int SEQ   = 2048;
constexpr int DM    = 1024;
constexpr int NH    = 16;
constexpr int DH    = 64;
constexpr float NEG_INF = -1000000000.0f;

// ---------- helpers ----------
DEV __bf16 f2bf(float f) { return (__bf16)f; }   // native cvt (RNE)

DEV v16bf combine16(v8bf lo, v8bf hi) {
  v16bf r;
#pragma unroll
  for (int i = 0; i < 8; ++i) { r[i] = lo[i]; r[i + 8] = hi[i]; }
  return r;
}

// A-operand (16x32 bf16) from row-major [rows, ld]:
// lane m (0-15): K = {k0..k0+7, k0+16..k0+23}; lane m+16: K = {+8..+15, +24..+31}
DEV v16bf load_a_bf(const __bf16* base, int ld, int row0, int k0, int lane) {
  int hl = lane >> 4, lm = lane & 15;
  const __bf16* p = base + (size_t)(row0 + lm) * ld + k0 + hl * 8;
  v8bf lo = *(const v8bf*)p;
  v8bf hi = *(const v8bf*)(p + 16);
  return combine16(lo, hi);
}

// B-operand (32x16 bf16) fed from transposed storage Bt[n, k] row-major:
// lane n (0-15): K = {k0..k0+7, k0+16..k0+23}; lane n+16: K = {+8..+15, +24..+31}
DEV v16bf load_b_bf(const __bf16* baseT, int ld, int n0, int k0, int lane) {
  int hl = lane >> 4, lm = lane & 15;
  const __bf16* p = baseT + (size_t)(n0 + lm) * ld + k0 + hl * 8;
  v8bf lo = *(const v8bf*)p;
  v8bf hi = *(const v8bf*)(p + 16);
  return combine16(lo, hi);
}

DEV v8f wmma_bf(v16bf a, v16bf b, v8f c) {
  return __builtin_amdgcn_wmma_f32_16x16x32_bf16(
      false, a, false, b, (short)0, c, false, false);
}

// ---------- kernel 1: activation fp32 -> bf16, vectorized (b128 in/out) ----------
__global__ __launch_bounds__(256) void xconv_kernel(const float* __restrict__ X,
                                                    __bf16* __restrict__ Xb) {
  int idx = blockIdx.x * 256 + threadIdx.x;
  const float4* p = (const float4*)X + (size_t)idx * 2;
  float4 a = p[0], b = p[1];
  v8bf o;
  o[0] = f2bf(a.x); o[1] = f2bf(a.y); o[2] = f2bf(a.z); o[3] = f2bf(a.w);
  o[4] = f2bf(b.x); o[5] = f2bf(b.y); o[6] = f2bf(b.z); o[7] = f2bf(b.w);
  ((v8bf*)Xb)[idx] = o;
}

// ---------- kernel 2: LDS-tiled weight transpose + convert: Wt[n,k]=W[k,n] ----------
__global__ __launch_bounds__(256) void wtrans_kernel(const float* __restrict__ W,
                                                     __bf16* __restrict__ Wt) {
  __shared__ float tile[32][33];
  int tx = threadIdx.x & 31;
  int ty = threadIdx.x >> 5;
  int bx = blockIdx.x & 31;
  int by = blockIdx.x >> 5;
#pragma unroll
  for (int i = 0; i < 4; ++i) {
    int k = by * 32 + ty + 8 * i;
    tile[ty + 8 * i][tx] = W[(size_t)k * DM + bx * 32 + tx];
  }
  __syncthreads();
#pragma unroll
  for (int i = 0; i < 4; ++i) {
    int n = bx * 32 + ty + 8 * i;
    Wt[(size_t)n * DM + by * 32 + tx] = f2bf(tile[tx][ty + 8 * i]);
  }
}

// ---------- kernel 3: GEMM  C[8192,1024] = A @ W + bias (32x64 tile / wave) ----------
// mode 0: bf16 head-split [B,H,S,D] ; mode 1: bf16 transposed [B,H,D,S] ;
// mode 2: fp32 row-major [M,DM]
__global__ __launch_bounds__(256) void gemm_kernel(
    const __bf16* __restrict__ A, const __bf16* __restrict__ Wt,
    const float* __restrict__ bias, __bf16* __restrict__ outb,
    float* __restrict__ outf, int mode) {
  int lane = threadIdx.x & 31;
  int w = blockIdx.x * 8 + (threadIdx.x >> 5);
  int nt = w & 15, mt = w >> 4;
  int m0 = mt * 32, n0 = nt * 64;

  v8f acc[2][4] = {};
  v16bf a0[2], a1[2], bb[2][4];   // ping-pong slots, constant-indexed

  auto load_slot = [&](int s, int k0) {
    a0[s] = load_a_bf(A, DM, m0, k0, lane);
    a1[s] = load_a_bf(A, DM, m0 + 16, k0, lane);
#pragma unroll
    for (int j = 0; j < 4; ++j) bb[s][j] = load_b_bf(Wt, DM, n0 + 16 * j, k0, lane);
  };
  auto do_wmmas = [&](int s) {
#pragma unroll
    for (int j = 0; j < 4; ++j) {
      acc[0][j] = wmma_bf(a0[s], bb[s][j], acc[0][j]);
      acc[1][j] = wmma_bf(a1[s], bb[s][j], acc[1][j]);
    }
  };

  load_slot(0, 0);
  for (int k0 = 0; k0 < DM - 64; k0 += 64) {      // 15 iterations, unroll-by-2 body
    load_slot(1, k0 + 32);
    do_wmmas(0);
    load_slot(0, k0 + 64);
    do_wmmas(1);
  }
  load_slot(1, DM - 32);                          // peeled tail
  do_wmmas(0);
  do_wmmas(1);

  int hl = lane >> 4, ln = lane & 15;
#pragma unroll
  for (int rb = 0; rb < 2; ++rb) {
#pragma unroll
    for (int j = 0; j < 4; ++j) {
      int col = n0 + 16 * j + ln;
      float bv = bias[col];
#pragma unroll
      for (int r = 0; r < 8; ++r) acc[rb][j][r] += bv;

      if (mode == 2) {
#pragma unroll
        for (int r = 0; r < 8; ++r) {
          int row = m0 + 16 * rb + r + 8 * hl;
          outf[(size_t)row * DM + col] = acc[rb][j][r];
        }
      } else if (mode == 0) {
        int h = col >> 6, d = col & 63;
#pragma unroll
        for (int r = 0; r < 8; ++r) {
          int row = m0 + 16 * rb + r + 8 * hl;
          int b = row >> 11, s = row & 2047;
          outb[(((size_t)(b * NH + h) * SEQ + s) << 6) + d] = f2bf(acc[rb][j][r]);
        }
      } else {  // mode 1: [B,H,D,S]; 8 consecutive s per lane -> b128 store
        int h = col >> 6, d = col & 63;
        int b = m0 >> 11;
        int s0 = (m0 & 2047) + 16 * rb + 8 * hl;
        v8bf pk;
#pragma unroll
        for (int r = 0; r < 8; ++r) pk[r] = f2bf(acc[rb][j][r]);
        *(v8bf*)(outb + ((size_t)(b * NH + h) * DH + d) * SEQ + s0) = pk;
      }
    }
  }
}

// ---------- kernel 4: flash attention, one (b,h,16-query tile) per wave ----------
__global__ __launch_bounds__(256) void attn_kernel(
    const __bf16* __restrict__ Qh, const __bf16* __restrict__ Kh,
    const __bf16* __restrict__ Vt, const int* __restrict__ mask,
    __bf16* __restrict__ AO) {
  __shared__ __bf16 plds[8][16 * 32];
  int lane = threadIdx.x & 31;
  int wv = threadIdx.x >> 5;
  int w = blockIdx.x * 8 + wv;
  int qt = w & 127;
  int bh = w >> 7;
  int b = bh >> 4, h = bh & 15;
  int q0 = qt * 16;

  const __bf16* Qb = Qh + (size_t)bh * SEQ * DH;
  const __bf16* Kb = Kh + (size_t)bh * SEQ * DH;
  const __bf16* Vb = Vt + (size_t)bh * DH * SEQ;
  const int* mk = mask + b * SEQ;

  v16bf aq0 = load_a_bf(Qb, DH, q0, 0, lane);
  v16bf aq1 = load_a_bf(Qb, DH, q0, 32, lane);

  v8f acc[4] = {};
  float mx[8], ls[8];
#pragma unroll
  for (int r = 0; r < 8; ++r) { mx[r] = -1e30f; ls[r] = 0.f; }
  int hl = lane >> 4, ln = lane & 15;
  __bf16* lp = plds[wv];

  v16bf bk[2][4];                 // double-buffered K tiles
  auto load_k = [&](int s, int kb) {
    bk[s][0] = load_b_bf(Kb, DH, kb, 0, lane);
    bk[s][1] = load_b_bf(Kb, DH, kb, 32, lane);
    bk[s][2] = load_b_bf(Kb, DH, kb + 16, 0, lane);
    bk[s][3] = load_b_bf(Kb, DH, kb + 16, 32, lane);
  };

  auto process = [&](int s, int kb) {
    v8f z = {};
    // V tiles + mask: consumed after softmax, latency hidden under the VALU work
    v16bf bv[4];
#pragma unroll
    for (int j = 0; j < 4; ++j) bv[j] = load_b_bf(Vb, SEQ, j * 16, kb, lane);
    float msk0 = (float)mk[kb + ln] * NEG_INF;
    float msk1 = (float)mk[kb + 16 + ln] * NEG_INF;

    v8f c0 = wmma_bf(aq0, bk[s][0], z);
    c0     = wmma_bf(aq1, bk[s][1], c0);
    v8f c1 = wmma_bf(aq0, bk[s][2], z);
    c1     = wmma_bf(aq1, bk[s][3], c1);

    float corr[8];
#pragma unroll
    for (int r = 0; r < 8; ++r) {
      float e0 = c0[r] * 0.125f + msk0;           // scale = 1/sqrt(64)
      float e1 = c1[r] * 0.125f + msk1;
      float rm = fmaxf(e0, e1);
      rm = fmaxf(rm, __shfl_xor(rm, 1, 32));
      rm = fmaxf(rm, __shfl_xor(rm, 2, 32));
      rm = fmaxf(rm, __shfl_xor(rm, 4, 32));
      rm = fmaxf(rm, __shfl_xor(rm, 8, 32));
      float mn = fmaxf(mx[r], rm);
      float cr = __expf(mx[r] - mn);
      float p0 = __expf(e0 - mn);
      float p1 = __expf(e1 - mn);
      float rs = p0 + p1;
      rs += __shfl_xor(rs, 1, 32);
      rs += __shfl_xor(rs, 2, 32);
      rs += __shfl_xor(rs, 4, 32);
      rs += __shfl_xor(rs, 8, 32);
      ls[r] = ls[r] * cr + rs;
      mx[r] = mn;
      corr[r] = cr;
      int m = r + 8 * hl;                         // C-layout -> LDS row-major 16x32
      lp[m * 32 + ln] = f2bf(p0);
      lp[m * 32 + 16 + ln] = f2bf(p1);
    }

#pragma unroll
    for (int j = 0; j < 4; ++j)
#pragma unroll
      for (int r = 0; r < 8; ++r) acc[j][r] *= corr[r];

    v16bf pA = load_a_bf(lp, 32, 0, 0, lane);     // A-layout readback (ds waits auto)
#pragma unroll
    for (int j = 0; j < 4; ++j) acc[j] = wmma_bf(pA, bv[j], acc[j]);
  };

  load_k(0, 0);
  for (int kb = 0; kb < SEQ - 64; kb += 64) {     // 31 iterations, unroll-by-2
    load_k(1, kb + 32);
    __builtin_prefetch(Kb + (size_t)(kb + 64) * DH, 0, 3);   // gfx1250 prefetch
    process(0, kb);
    load_k(0, kb + 64);
    process(1, kb + 32);
  }
  load_k(1, SEQ - 32);                            // peeled tail
  process(0, SEQ - 64);
  process(1, SEQ - 32);

#pragma unroll
  for (int j = 0; j < 4; ++j) {
    int col = h * DH + j * 16 + ln;
#pragma unroll
    for (int r = 0; r < 8; ++r) {
      int srow = b * SEQ + q0 + r + 8 * hl;
      AO[(size_t)srow * DM + col] = f2bf(acc[j][r] / ls[r]);
    }
  }
}

// ---------- host ----------
extern "C" void kernel_launch(void* const* d_in, const int* in_sizes, int n_in,
                              void* d_out, int out_size, void* d_ws, size_t ws_size,
                              hipStream_t stream) {
  (void)in_sizes; (void)n_in; (void)out_size; (void)ws_size;
  const float* q    = (const float*)d_in[0];
  const float* k    = (const float*)d_in[1];
  const float* v    = (const float*)d_in[2];
  const int*   mask = (const int*)d_in[3];
  const float* Wq   = (const float*)d_in[4];
  const float* bq   = (const float*)d_in[5];
  const float* Wk   = (const float*)d_in[6];
  const float* bk   = (const float*)d_in[7];
  const float* Wv   = (const float*)d_in[8];
  const float* bv   = (const float*)d_in[9];
  const float* Wo   = (const float*)d_in[10];
  const float* bo   = (const float*)d_in[11];
  float* out = (float*)d_out;

  // workspace layout (88 MB total)
  char* ws = (char*)d_ws;
  __bf16* Wtq = (__bf16*)(ws + (size_t)0);
  __bf16* Wtk = (__bf16*)(ws + ((size_t)2 << 20));
  __bf16* Wtv = (__bf16*)(ws + ((size_t)4 << 20));
  __bf16* Wto = (__bf16*)(ws + ((size_t)6 << 20));
  __bf16* Xb  = (__bf16*)(ws + ((size_t)8 << 20));   // reused per projection
  __bf16* Qh  = (__bf16*)(ws + ((size_t)24 << 20));  // [B,H,S,D]
  __bf16* Kh  = (__bf16*)(ws + ((size_t)40 << 20));  // [B,H,S,D]
  __bf16* Vt  = (__bf16*)(ws + ((size_t)56 << 20));  // [B,H,D,S]
  __bf16* AO  = (__bf16*)(ws + ((size_t)72 << 20));  // [B*S, DM]

  wtrans_kernel<<<1024, 256, 0, stream>>>(Wq, Wtq);
  wtrans_kernel<<<1024, 256, 0, stream>>>(Wk, Wtk);
  wtrans_kernel<<<1024, 256, 0, stream>>>(Wv, Wtv);
  wtrans_kernel<<<1024, 256, 0, stream>>>(Wo, Wto);

  xconv_kernel<<<4096, 256, 0, stream>>>(q, Xb);
  gemm_kernel<<<512, 256, 0, stream>>>(Xb, Wtq, bq, Qh, nullptr, 0);
  xconv_kernel<<<4096, 256, 0, stream>>>(k, Xb);
  gemm_kernel<<<512, 256, 0, stream>>>(Xb, Wtk, bk, Kh, nullptr, 0);
  xconv_kernel<<<4096, 256, 0, stream>>>(v, Xb);
  gemm_kernel<<<512, 256, 0, stream>>>(Xb, Wtv, bv, Vt, nullptr, 1);

  attn_kernel<<<1024, 256, 0, stream>>>(Qh, Kh, Vt, mask, AO);

  gemm_kernel<<<512, 256, 0, stream>>>(AO, Wto, bo, nullptr, out, 2);
}
